// LanguageModel_63428077027722
// MI455X (gfx1250) — matrix operand order repaired and checked by reading, not compiled, not run on hardware
//
#include <hip/hip_runtime.h>

// GRU language-model decoder for MI455X (gfx1250, wave32, WMMA).
// All GEMMs run on v_wmma_f32_16x16x32_f16 (f16 in, f32 accumulate).
// Weights (~4MB) are L2-resident; recurrence = 14 sequential step kernels.
// Wave tile = 32(M) x 32(N). All fragment base pointers are hoisted out of
// the K loops so every load is global_load_b128 [base] offset:imm.

typedef _Float16 f16;
typedef __attribute__((ext_vector_type(16))) _Float16 v16h;
typedef __attribute__((ext_vector_type(8)))  _Float16 v8h;
typedef __attribute__((ext_vector_type(8)))  float    v8f;

#define BATCH   8192
#define EMBED   256
#define HIDDEN  512
#define VOCAB   193
#define TSTEPS  14
#define MAXLEN  15
#define H3      (3 * HIDDEN)
#define NPAD    224   // W_out rows padded to 7 x 32-col tiles

// ---------------------------------------------------------------------------
// WMMA helpers
// ---------------------------------------------------------------------------
__device__ inline v16h cat8(v8h lo, v8h hi) {
  return __builtin_shufflevector(lo, hi, 0,1,2,3,4,5,6,7,8,9,10,11,12,13,14,15);
}

// A fragment at a precomputed lane base pointer (row-major A):
// lane L: row = m0 + L%16, contiguous k-range (L/16)*16 + [0..15].
// base = A + (m0 + L%16)*lda + (L/16)*16;  load at base + kc and + kc + 8.
__device__ inline v16h load_a(const f16* __restrict__ p) {
  v8h lo = *(const v8h*)(p);
  v8h hi = *(const v8h*)(p + 8);
  return cat8(lo, hi);
}

// B fragment (B column n = W row n, W row-major [N,K]):
// lane L: col = n0 + L%16, contiguous k-range (L/16)*16 + [0..15].
// base = W + (n0 + L%16)*ldb + (L/16)*16;  load at base + kc and + kc + 8.
__device__ inline v16h load_b(const f16* __restrict__ p) {
  v8h lo = *(const v8h*)(p);
  v8h hi = *(const v8h*)(p + 8);
  return cat8(lo, hi);
}

__device__ inline v8f wmma_f16(v16h a, v16h b, v8f c) {
  return __builtin_amdgcn_wmma_f32_16x16x32_f16(
      /*neg_a=*/false, a, /*neg_b=*/false, b,
      /*c_mod=*/(short)0, c, /*reuse_a=*/false, /*reuse_b=*/false);
}

__device__ inline float fsigmoid(float x) { return 1.0f / (1.0f + __expf(-x)); }
__device__ inline float ftanh(float x) {
  x = fminf(fmaxf(x, -15.0f), 15.0f);
  float e = __expf(-2.0f * x);
  return (1.0f - e) / (1.0f + e);
}

// ---------------------------------------------------------------------------
// One-time prep kernels
// ---------------------------------------------------------------------------
__global__ void cvt_f32_f16(const float* __restrict__ src, f16* __restrict__ dst, int n) {
  int i = blockIdx.x * 256 + threadIdx.x;
  if (i < n) dst[i] = (f16)src[i];
}

__global__ void pad_wout(const float* __restrict__ src, f16* __restrict__ dst) {
  int i = blockIdx.x * 256 + threadIdx.x;          // over NPAD*HIDDEN
  if (i >= NPAD * HIDDEN) return;
  dst[i] = (i < VOCAB * HIDDEN) ? (f16)src[i] : (f16)0.0f;
}

// X_all[t][b][:] = emb_h[text[b][t]][:]  (16B vector copies)
__global__ void gather_emb(const int* __restrict__ text,
                           const f16* __restrict__ embh,
                           f16* __restrict__ xall) {
  int idx = blockIdx.x * 256 + threadIdx.x;        // TSTEPS*BATCH*32 threads
  if (idx >= TSTEPS * BATCH * (EMBED / 8)) return;
  int e8 = idx & 31;
  int b  = (idx >> 5) & (BATCH - 1);
  int t  = idx >> 18;
  int tok = text[b * MAXLEN + t];
  const v8h* s = (const v8h*)(embh + (size_t)tok * EMBED) + e8;
  v8h* d = (v8h*)(xall + ((size_t)t * BATCH + b) * EMBED) + e8;
  *d = *s;
}

// ---------------------------------------------------------------------------
// Generic WMMA GEMM (K = HIDDEN = 512): C[m,n] = A[m,:]·W[n,:] + bias[n]
// Block = 256 thr = 8 waves; wave tile 32(M) x 32(N); block tile 256 x 32.
// MODE 0: write f32 out (row stride HIDDEN) and f16 mirror (h0 path).
// MODE 1: output head: m = t*BATCH+b -> d_out[(b*T+t)*V + n], guard n<VOCAB.
// ---------------------------------------------------------------------------
template <int MODE>
__global__ __launch_bounds__(256) void gemm_wmma(
    const f16* __restrict__ A, const f16* __restrict__ W,
    const float* __restrict__ bias,
    float* __restrict__ outF, f16* __restrict__ outH) {
  const int lane = threadIdx.x & 31;
  const int wave = threadIdx.x >> 5;
  const int m0 = blockIdx.x * 256 + wave * 32;
  const int n0 = blockIdx.y * 32;
  const int r = lane & 15, half = lane >> 4;

  // stationary lane base pointers (K-loop offsets become immediates)
  const f16* pA0 = A + (size_t)(m0 + r) * HIDDEN + half * 16;
  const f16* pA1 = pA0 + (size_t)16 * HIDDEN;
  const f16* pB0 = W + (size_t)(n0 + r) * HIDDEN + half * 16;
  const f16* pB1 = pB0 + (size_t)16 * HIDDEN;

  v8f acc[2][2];   // [mt][nt]
#pragma unroll
  for (int mt = 0; mt < 2; ++mt)
#pragma unroll
    for (int nt = 0; nt < 2; ++nt) acc[mt][nt] = (v8f)0.0f;

#pragma unroll
  for (int kc = 0; kc < HIDDEN; kc += 32) {
    v16h a0 = load_a(pA0 + kc);
    v16h a1 = load_a(pA1 + kc);
    v16h b0 = load_b(pB0 + kc);
    v16h b1 = load_b(pB1 + kc);
    acc[0][0] = wmma_f16(a0, b0, acc[0][0]);
    acc[0][1] = wmma_f16(a0, b1, acc[0][1]);
    acc[1][0] = wmma_f16(a1, b0, acc[1][0]);
    acc[1][1] = wmma_f16(a1, b1, acc[1][1]);
  }

#pragma unroll
  for (int nt = 0; nt < 2; ++nt) {
    const int n = n0 + nt * 16 + r;
    float bv = 0.0f;
    if (MODE == 0) bv = bias[n];
    else if (n < VOCAB) bv = bias[n];
#pragma unroll
    for (int mt = 0; mt < 2; ++mt) {
#pragma unroll
      for (int i = 0; i < 8; ++i) {
        const int m = m0 + mt * 16 + half * 8 + i;
        const float v = acc[mt][nt][i] + bv;
        if (MODE == 0) {
          size_t idx = (size_t)m * HIDDEN + n;
          outF[idx] = v;
          outH[idx] = (f16)v;
        } else {
          if (n < VOCAB) {
            int t = m >> 13;               // m / BATCH
            int b = m & (BATCH - 1);
            outF[((size_t)b * TSTEPS + t) * VOCAB + n] = v;
          }
        }
      }
    }
  }
}

// ---------------------------------------------------------------------------
// One GRU step. Wave tile: 32 rows x 32 h-columns, four accumulator groups
// (R, Z combined input+hidden; N kept split so r can gate only the hidden
// part). K loop: 256 over x_t/W_ih then 512 over h/W_hh. Gates fused.
// ---------------------------------------------------------------------------
__global__ __launch_bounds__(256) void gru_step(
    const f16* __restrict__ Xt,     // [BATCH, EMBED]
    const f16* __restrict__ Hprev,  // [BATCH, HIDDEN] f16
    const f16* __restrict__ Wih,    // [3H, EMBED]
    const f16* __restrict__ Whh,    // [3H, HIDDEN]
    const float* __restrict__ b_ih, const float* __restrict__ b_hh,
    float* __restrict__ Hf32,       // [BATCH, HIDDEN] in/out (per-elem RMW)
    f16* __restrict__ Hout) {       // [BATCH, HIDDEN] f16 (= hs[t])
  const int lane = threadIdx.x & 31;
  const int wave = threadIdx.x >> 5;
  const int m0 = blockIdx.x * 256 + wave * 32;
  const int n0 = blockIdx.y * 32;
  const int r = lane & 15, half = lane >> 4;

  v8f aR[2][2], aZ[2][2], aIN[2][2], aHN[2][2];   // [mt][nt]
#pragma unroll
  for (int mt = 0; mt < 2; ++mt)
#pragma unroll
    for (int nt = 0; nt < 2; ++nt) {
      aR[mt][nt] = (v8f)0.0f; aZ[mt][nt] = (v8f)0.0f;
      aIN[mt][nt] = (v8f)0.0f; aHN[mt][nt] = (v8f)0.0f;
    }

  // ---- input contribution: gi = x_t @ W_ih^T (K = 256) -------------------
  {
    const f16* pA0 = Xt + (size_t)(m0 + r) * EMBED + half * 16;
    const f16* pA1 = pA0 + (size_t)16 * EMBED;
    const f16* pW[2][3];
#pragma unroll
    for (int nt = 0; nt < 2; ++nt) {
      const int nj = n0 + nt * 16 + r;
#pragma unroll
      for (int g = 0; g < 3; ++g)
        pW[nt][g] = Wih + (size_t)(g * HIDDEN + nj) * EMBED + half * 16;
    }
#pragma unroll
    for (int kc = 0; kc < EMBED; kc += 32) {
      v16h a0 = load_a(pA0 + kc);
      v16h a1 = load_a(pA1 + kc);
#pragma unroll
      for (int nt = 0; nt < 2; ++nt) {
        v16h bR = load_b(pW[nt][0] + kc);
        v16h bZ = load_b(pW[nt][1] + kc);
        v16h bN = load_b(pW[nt][2] + kc);
        aR[0][nt]  = wmma_f16(a0, bR, aR[0][nt]);
        aR[1][nt]  = wmma_f16(a1, bR, aR[1][nt]);
        aZ[0][nt]  = wmma_f16(a0, bZ, aZ[0][nt]);
        aZ[1][nt]  = wmma_f16(a1, bZ, aZ[1][nt]);
        aIN[0][nt] = wmma_f16(a0, bN, aIN[0][nt]);
        aIN[1][nt] = wmma_f16(a1, bN, aIN[1][nt]);
      }
    }
  }

  // ---- hidden contribution: gh = h @ W_hh^T (K = 512) --------------------
  {
    const f16* pA0 = Hprev + (size_t)(m0 + r) * HIDDEN + half * 16;
    const f16* pA1 = pA0 + (size_t)16 * HIDDEN;
    const f16* pW[2][3];
#pragma unroll
    for (int nt = 0; nt < 2; ++nt) {
      const int nj = n0 + nt * 16 + r;
#pragma unroll
      for (int g = 0; g < 3; ++g)
        pW[nt][g] = Whh + (size_t)(g * HIDDEN + nj) * HIDDEN + half * 16;
    }
#pragma unroll
    for (int kc = 0; kc < HIDDEN; kc += 32) {
      v16h a0 = load_a(pA0 + kc);
      v16h a1 = load_a(pA1 + kc);
#pragma unroll
      for (int nt = 0; nt < 2; ++nt) {
        v16h bR = load_b(pW[nt][0] + kc);
        v16h bZ = load_b(pW[nt][1] + kc);
        v16h bN = load_b(pW[nt][2] + kc);
        aR[0][nt]  = wmma_f16(a0, bR, aR[0][nt]);
        aR[1][nt]  = wmma_f16(a1, bR, aR[1][nt]);
        aZ[0][nt]  = wmma_f16(a0, bZ, aZ[0][nt]);
        aZ[1][nt]  = wmma_f16(a1, bZ, aZ[1][nt]);
        aHN[0][nt] = wmma_f16(a0, bN, aHN[0][nt]);
        aHN[1][nt] = wmma_f16(a1, bN, aHN[1][nt]);
      }
    }
  }

  // ---- fused gate epilogue ----------------------------------------------
#pragma unroll
  for (int nt = 0; nt < 2; ++nt) {
    const int j = n0 + nt * 16 + r;
    const float bR  = b_ih[j] + b_hh[j];
    const float bZ  = b_ih[HIDDEN + j] + b_hh[HIDDEN + j];
    const float bNi = b_ih[2 * HIDDEN + j];
    const float bNh = b_hh[2 * HIDDEN + j];
#pragma unroll
    for (int mt = 0; mt < 2; ++mt) {
#pragma unroll
      for (int i = 0; i < 8; ++i) {
        const int m = m0 + mt * 16 + half * 8 + i;
        const float rg = fsigmoid(aR[mt][nt][i] + bR);
        const float zg = fsigmoid(aZ[mt][nt][i] + bZ);
        const float ng = ftanh(aIN[mt][nt][i] + bNi + rg * (aHN[mt][nt][i] + bNh));
        const size_t idx = (size_t)m * HIDDEN + j;
        const float hp = Hf32[idx];                 // same thread reads+writes
        const float hn = (1.0f - zg) * ng + zg * hp;
        Hf32[idx] = hn;
        Hout[idx] = (f16)hn;
      }
    }
  }
}

// ---------------------------------------------------------------------------
extern "C" void kernel_launch(void* const* d_in, const int* in_sizes, int n_in,
                              void* d_out, int out_size, void* d_ws, size_t ws_size,
                              hipStream_t stream) {
  (void)in_sizes; (void)n_in; (void)out_size; (void)ws_size;
  const float* img    = (const float*)d_in[0];
  const int*   text   = (const int*)  d_in[1];
  const float* emb    = (const float*)d_in[2];
  const float* W_proj = (const float*)d_in[3];
  const float* b_proj = (const float*)d_in[4];
  const float* W_ih   = (const float*)d_in[5];
  const float* W_hh   = (const float*)d_in[6];
  const float* b_ih   = (const float*)d_in[7];
  const float* b_hh   = (const float*)d_in[8];
  const float* W_out  = (const float*)d_in[9];
  const float* b_out  = (const float*)d_in[10];
  float* out = (float*)d_out;

  // workspace carve-up (256B aligned); total ~213 MB
  char* ws = (char*)d_ws;
  size_t o = 0;
  auto carve = [&](size_t bytes) -> char* {
    char* p = ws + o;
    o = (o + bytes + 255) & ~(size_t)255;
    return p;
  };
  f16* wproj_h = (f16*)carve((size_t)HIDDEN * HIDDEN * 2);
  f16* wih_h   = (f16*)carve((size_t)H3 * EMBED * 2);
  f16* whh_h   = (f16*)carve((size_t)H3 * HIDDEN * 2);
  f16* wout_h  = (f16*)carve((size_t)NPAD * HIDDEN * 2);
  f16* emb_h   = (f16*)carve((size_t)VOCAB * EMBED * 2);
  f16* img_h   = (f16*)carve((size_t)BATCH * HIDDEN * 2);
  f16* xall    = (f16*)carve((size_t)TSTEPS * BATCH * EMBED * 2);
  float* hf32  = (float*)carve((size_t)BATCH * HIDDEN * 4);
  f16* hseq    = (f16*)carve((size_t)(TSTEPS + 1) * BATCH * HIDDEN * 2);

  auto cdiv = [](int a, int b) { return (a + b - 1) / b; };

  // 1) weight / embedding / activation conversion to f16
  cvt_f32_f16<<<cdiv(HIDDEN * HIDDEN, 256), 256, 0, stream>>>(W_proj, wproj_h, HIDDEN * HIDDEN);
  cvt_f32_f16<<<cdiv(H3 * EMBED, 256), 256, 0, stream>>>(W_ih, wih_h, H3 * EMBED);
  cvt_f32_f16<<<cdiv(H3 * HIDDEN, 256), 256, 0, stream>>>(W_hh, whh_h, H3 * HIDDEN);
  pad_wout<<<cdiv(NPAD * HIDDEN, 256), 256, 0, stream>>>(W_out, wout_h);
  cvt_f32_f16<<<cdiv(VOCAB * EMBED, 256), 256, 0, stream>>>(emb, emb_h, VOCAB * EMBED);
  cvt_f32_f16<<<cdiv(BATCH * HIDDEN, 256), 256, 0, stream>>>(img, img_h, BATCH * HIDDEN);

  // 2) embedding gather: X_all[t][b] = emb[text[b][t]]
  gather_emb<<<cdiv(TSTEPS * BATCH * (EMBED / 8), 256), 256, 0, stream>>>(text, emb_h, xall);

  // 3) h0 = img @ W_proj^T + b_proj  -> hf32 (f32) and hseq[0] (f16)
  gemm_wmma<0><<<dim3(BATCH / 256, HIDDEN / 32), 256, 0, stream>>>(
      img_h, wproj_h, b_proj, hf32, hseq);

  // 4) 14 sequential GRU steps (kernel boundary = global sync of recurrence)
  for (int t = 0; t < TSTEPS; ++t) {
    const f16* xt = xall + (size_t)t * BATCH * EMBED;
    const f16* hp = hseq + (size_t)t * BATCH * HIDDEN;
    f16* ho = hseq + (size_t)(t + 1) * BATCH * HIDDEN;   // doubles as hs[t]
    gru_step<<<dim3(BATCH / 256, HIDDEN / 32), 256, 0, stream>>>(
        xt, hp, wih_h, whh_h, b_ih, b_hh, hf32, ho);
  }

  // 5) preds = hs @ W_out^T + b_out, stored as [B, T, V]
  const f16* hs = hseq + (size_t)BATCH * HIDDEN;          // [T*B, H]
  gemm_wmma<1><<<dim3((TSTEPS * BATCH) / 256, NPAD / 32), 256, 0, stream>>>(
      hs, wout_h, b_out, out, nullptr);
}
